// DeepFakeDetectionModel_6640019440056
// MI455X (gfx1250) — compile-verified, matrix operand order checked by reading, not run on hardware
//
#include <hip/hip_runtime.h>

typedef _Float16 v8h  __attribute__((ext_vector_type(8)));
typedef _Float16 v16h __attribute__((ext_vector_type(16)));
typedef float    v8f  __attribute__((ext_vector_type(8)));

#define HDIM    1024
#define RS      1032      // padded LDS row stride in halves (2064 B: 16B-aligned, bank ~= 4*row)
#define BM      64        // batch rows per workgroup
#define NTHREADS 512      // 16 wave32s
#define NLAYERS 8
#define FIN     26
// 2 activation buffers + 3 small reduction scratch arrays (s1, s2, dot)
#define ACT_HALVES   (BM * RS)
#define PART_FLOATS  (BM * 8)
#define SMEM_BYTES   (2 * ACT_HALVES * (int)sizeof(_Float16) + 3 * PART_FLOATS * (int)sizeof(float))

// ---- fragment loader: lane-local 16 halves = [k..k+7] and [k+16..k+23] ----
__device__ __forceinline__ v16h load_frag16(const _Float16* p) {
  v8h lo = *(const v8h*)p;
  v8h hi = *(const v8h*)(p + 16);
  return __builtin_shufflevector(lo, hi, 0,1,2,3,4,5,6,7,8,9,10,11,12,13,14,15);
}

// ---- store a 16x16 f32 accumulator tile to LDS with ReLU, f16 ----
__device__ __forceinline__ void store_tile_relu(_Float16* buf, int mtile, int n0,
                                                int lane, const v8f& a) {
  const int lm    = lane & 15;
  const int rbase = mtile * 16 + ((lane >> 4) ? 8 : 0);
  const int col   = n0 + lm;
#pragma unroll
  for (int r = 0; r < 8; ++r) {
    float v = a[r];
    v = v > 0.f ? v : 0.f;
    buf[(rbase + r) * RS + col] = (_Float16)v;
  }
}

// ================= weight conversion kernels (f32 -> f16, run each call) =================
__global__ void cvt_blk_kernel(const float* __restrict__ w, _Float16* __restrict__ o, int n) {
  int i = blockIdx.x * blockDim.x + threadIdx.x;
  if (i < n) o[i] = (_Float16)w[i];
}

__global__ void cvt_stem_kernel(const float* __restrict__ w, _Float16* __restrict__ o) {
  int i = blockIdx.x * blockDim.x + threadIdx.x;   // 0 .. 1024*32-1
  if (i >= HDIM * 32) return;
  int row = i >> 5, col = i & 31;
  o[i] = (col < FIN) ? (_Float16)w[row * FIN + col] : (_Float16)0.f;
}

// ================= fused MLP kernel =================
__global__ void __launch_bounds__(NTHREADS, 1)
mlp_fused_kernel(const float* __restrict__ x,
                 const float* __restrict__ st_gamma, const float* __restrict__ st_beta,
                 const float* __restrict__ st_b,
                 const float* __restrict__ blk_gamma, const float* __restrict__ blk_beta,
                 const float* __restrict__ blk_b,
                 const float* __restrict__ last_gamma, const float* __restrict__ last_beta,
                 const float* __restrict__ last_w, const float* __restrict__ last_b,
                 const _Float16* __restrict__ stemW16, const _Float16* __restrict__ blkW16,
                 float* __restrict__ out) {
  extern __shared__ __align__(16) _Float16 smem[];
  _Float16* bufN = smem;                 // GEMM input  (LN output)   [BM][RS]
  _Float16* bufR = smem + ACT_HALVES;    // GEMM output (ReLU)        [BM][RS]
  float* p1 = (float*)(smem + 2 * ACT_HALVES);  // [BM][8] partial sums
  float* p2 = p1 + PART_FLOATS;                  // [BM][8] partial sq-sums
  float* p3 = p2 + PART_FLOATS;                  // [BM][8] partial dots (head)

  const int tid     = threadIdx.x;
  const int lane    = tid & 31;
  const int wave    = tid >> 5;          // 0..15
  const int rowBase = blockIdx.x * BM;

  const int mtile = wave & 3;            // which 16-row M tile (4 tiles)
  const int ntb   = (wave >> 2) * 16;    // base N-tile (16 tiles = 256 cols per wave)
  const int lm    = lane & 15;
  const int koff  = (lane >> 4) << 3;    // hi half-wave reads K+8

  // LN work mapping: one row per lane (32 distinct rows per wave -> ~2-way LDS conflicts max)
  const int lrow = tid & 63;             // row handled by this thread
  const int lseg = tid >> 6;             // which 128-col segment (0..7)
  const int c0   = lseg * 128;

  // ---------------- stem LayerNorm over 26 features ----------------
  if (tid < BM) {
    const float* xr = x + (size_t)(rowBase + tid) * FIN;
    float v[FIN];
    float s1 = 0.f;
#pragma unroll
    for (int i = 0; i < FIN; ++i) { v[i] = xr[i]; s1 += v[i]; }
    const float m = s1 * (1.f / FIN);
    float s2 = 0.f;
#pragma unroll
    for (int i = 0; i < FIN; ++i) { float d = v[i] - m; s2 += d * d; }
    const float inv = rsqrtf(s2 * (1.f / FIN) + 1e-5f);
    _Float16* dst = bufN + tid * RS;
#pragma unroll
    for (int i = 0; i < FIN; ++i)
      dst[i] = (_Float16)((v[i] - m) * inv * st_gamma[i] + st_beta[i]);
#pragma unroll
    for (int i = FIN; i < 32; ++i) dst[i] = (_Float16)0.f;
  }
  __syncthreads();

  // ---------------- stem GEMM: [64 x 32] x [32 x 1024], one WMMA K step ----------------
  for (int ng = 0; ng < 16; ng += 8) {
    const int n0 = (ntb + ng) * 16;
    const v16h aF = load_frag16(bufN + (mtile * 16 + lm) * RS + koff);
    v8f acc[8];
#pragma unroll
    for (int t = 0; t < 8; ++t) {
      const float bv = st_b[n0 + t * 16 + lm];
#pragma unroll
      for (int r = 0; r < 8; ++r) acc[t][r] = bv;
      const v16h bF = load_frag16(stemW16 + (size_t)(n0 + t * 16 + lm) * 32 + koff);
      acc[t] = __builtin_amdgcn_wmma_f32_16x16x32_f16(false, aF, false, bF,
                                                      (short)0, acc[t], false, false);
    }
#pragma unroll
    for (int t = 0; t < 8; ++t) store_tile_relu(bufR, mtile, n0 + t * 16, lane, acc[t]);
  }

  // ---------------- 8 blocks: LN -> Linear(H->H) -> ReLU ----------------
  for (int l = 0; l < NLAYERS; ++l) {
    __syncthreads();
    // ---- LN stage 1: partial sums of bufR row ----
    {
      const _Float16* src = bufR + lrow * RS + c0;
      float s1 = 0.f, s2 = 0.f;
#pragma unroll
      for (int cc = 0; cc < 128; cc += 8) {
        v8h h = *(const v8h*)(src + cc);
#pragma unroll
        for (int i = 0; i < 8; ++i) { float v = (float)h[i]; s1 += v; s2 += v * v; }
      }
      p1[lrow * 8 + lseg] = s1;
      p2[lrow * 8 + lseg] = s2;
    }
    __syncthreads();
    // ---- LN stage 2: finish stats (redundantly per segment-thread), normalize ----
    {
      float s1 = 0.f, s2 = 0.f;
#pragma unroll
      for (int j = 0; j < 8; ++j) { s1 += p1[lrow * 8 + j]; s2 += p2[lrow * 8 + j]; }
      const float m   = s1 * (1.f / HDIM);
      const float var = s2 * (1.f / HDIM) - m * m;
      const float inv = rsqrtf(var + 1e-5f);
      const float* g  = blk_gamma + (size_t)l * HDIM + c0;
      const float* be = blk_beta  + (size_t)l * HDIM + c0;
      const _Float16* src = bufR + lrow * RS + c0;
      _Float16* dst = bufN + lrow * RS + c0;
#pragma unroll
      for (int cc = 0; cc < 128; cc += 8) {
        v8h h = *(const v8h*)(src + cc);
        v8h o;
#pragma unroll
        for (int i = 0; i < 8; ++i)
          o[i] = (_Float16)(((float)h[i] - m) * inv * g[cc + i] + be[cc + i]);
        *(v8h*)(dst + cc) = o;
      }
    }
    __syncthreads();
    // ---- GEMM bufN x W_l^T -> ReLU -> bufR ----
    const _Float16* wl = blkW16 + (size_t)l * HDIM * HDIM;
    const float*    bl = blk_b + (size_t)l * HDIM;
    for (int ng = 0; ng < 16; ng += 8) {
      const int n0 = (ntb + ng) * 16;
      v8f acc[8];
#pragma unroll
      for (int t = 0; t < 8; ++t) {
        const float bv = bl[n0 + t * 16 + lm];
#pragma unroll
        for (int r = 0; r < 8; ++r) acc[t][r] = bv;
      }
      const _Float16* arow = bufN + (mtile * 16 + lm) * RS + koff;
      for (int k = 0; k < HDIM; k += 32) {
        const v16h aF = load_frag16(arow + k);
        if (k + 32 < HDIM)
          __builtin_prefetch(wl + (size_t)(n0 + lm) * HDIM + k + 32 + koff, 0, 0);
#pragma unroll
        for (int t = 0; t < 8; ++t) {
          const v16h bF = load_frag16(wl + (size_t)(n0 + t * 16 + lm) * HDIM + k + koff);
          acc[t] = __builtin_amdgcn_wmma_f32_16x16x32_f16(false, aF, false, bF,
                                                          (short)0, acc[t], false, false);
        }
      }
#pragma unroll
      for (int t = 0; t < 8; ++t) store_tile_relu(bufR, mtile, n0 + t * 16, lane, acc[t]);
    }
  }
  __syncthreads();

  // ---------------- head: LN -> Linear(H->1) -> sigmoid ----------------
  {
    const _Float16* src = bufR + lrow * RS + c0;
    float s1 = 0.f, s2 = 0.f;
#pragma unroll
    for (int cc = 0; cc < 128; cc += 8) {
      v8h h = *(const v8h*)(src + cc);
#pragma unroll
      for (int i = 0; i < 8; ++i) { float v = (float)h[i]; s1 += v; s2 += v * v; }
    }
    p1[lrow * 8 + lseg] = s1;
    p2[lrow * 8 + lseg] = s2;
  }
  __syncthreads();
  {
    float s1 = 0.f, s2 = 0.f;
#pragma unroll
    for (int j = 0; j < 8; ++j) { s1 += p1[lrow * 8 + j]; s2 += p2[lrow * 8 + j]; }
    const float m   = s1 * (1.f / HDIM);
    const float var = s2 * (1.f / HDIM) - m * m;
    const float inv = rsqrtf(var + 1e-5f);
    const _Float16* src = bufR + lrow * RS + c0;
    const float* lg = last_gamma + c0;
    const float* lb = last_beta + c0;
    const float* lw = last_w + c0;
    float dot = 0.f;
#pragma unroll
    for (int cc = 0; cc < 128; cc += 8) {
      v8h h = *(const v8h*)(src + cc);
#pragma unroll
      for (int i = 0; i < 8; ++i) {
        float v = ((float)h[i] - m) * inv * lg[cc + i] + lb[cc + i];
        dot += v * lw[cc + i];
      }
    }
    p3[lrow * 8 + lseg] = dot;
  }
  __syncthreads();
  if (tid < BM) {
    float dot = 0.f;
#pragma unroll
    for (int j = 0; j < 8; ++j) dot += p3[tid * 8 + j];
    const float z = dot + last_b[0];
    out[rowBase + tid] = 1.f / (1.f + __expf(-z));
  }
}

// ================= host launcher =================
extern "C" void kernel_launch(void* const* d_in, const int* in_sizes, int n_in,
                              void* d_out, int out_size, void* d_ws, size_t ws_size,
                              hipStream_t stream) {
  (void)in_sizes; (void)n_in; (void)out_size; (void)ws_size;

  const float* x         = (const float*)d_in[0];
  const float* st_gamma  = (const float*)d_in[1];
  const float* st_beta   = (const float*)d_in[2];
  const float* st_w      = (const float*)d_in[3];
  const float* st_b      = (const float*)d_in[4];
  const float* blk_gamma = (const float*)d_in[5];
  const float* blk_beta  = (const float*)d_in[6];
  const float* blk_w     = (const float*)d_in[7];
  const float* blk_b     = (const float*)d_in[8];
  const float* last_gamma= (const float*)d_in[9];
  const float* last_beta = (const float*)d_in[10];
  const float* last_w    = (const float*)d_in[11];
  const float* last_b    = (const float*)d_in[12];
  float* out = (float*)d_out;

  // ws layout: stem weights padded to [1024][32] f16, then block weights [8][1024][1024] f16
  _Float16* stemW16 = (_Float16*)d_ws;
  _Float16* blkW16  = stemW16 + (size_t)HDIM * 32;

  // convert weights to f16 (every call; deterministic)
  {
    const int n = NLAYERS * HDIM * HDIM;
    cvt_blk_kernel<<<(n + 255) / 256, 256, 0, stream>>>(blk_w, blkW16, n);
    cvt_stem_kernel<<<(HDIM * 32 + 255) / 256, 256, 0, stream>>>(st_w, stemW16);
  }

  static_assert(SMEM_BYTES == 270336, "LDS budget (<= 320KB/WGP)");
  hipFuncSetAttribute((const void*)mlp_fused_kernel,
                      hipFuncAttributeMaxDynamicSharedMemorySize, SMEM_BYTES);

  const int nBlocks = 65536 / BM;  // 1024
  mlp_fused_kernel<<<nBlocks, NTHREADS, SMEM_BYTES, stream>>>(
      x, st_gamma, st_beta, st_b,
      blk_gamma, blk_beta, blk_b,
      last_gamma, last_beta, last_w, last_b,
      stemW16, blkW16, out);
}
